// SCAESuite_29231547416881
// MI455X (gfx1250) — matrix operand order repaired and checked by reading, not compiled, not run on hardware
//
#include <hip/hip_runtime.h>

typedef __bf16 bf16_t;
typedef __attribute__((ext_vector_type(16))) __bf16 v16bf;
typedef __attribute__((ext_vector_type(8)))  float  v8f;
typedef __attribute__((ext_vector_type(4)))  unsigned int u32x4;

union FragBF { u32x4 q[2]; v16bf v; };

static constexpr int Bc = 2, Sc = 1024, Dc = 512, Hc = 8, Fc = 3072, TOPK = 64;
static constexpr int BS = Bc * Sc;   // 2048

// 32-bit LDS byte offset of a generic pointer that points into LDS
// (AS3 pointers are offsets from the wave's LDS base, which is exactly what
//  GLOBAL_LOAD_ASYNC_TO_LDS expects in its VDST VGPR).
__device__ __forceinline__ unsigned lds_addr32(const void* p) {
  return (unsigned)(unsigned long long)(__attribute__((address_space(3))) const char*)p;
}

// ======================= elementwise / prep kernels =======================

__global__ void k_prep_act(const float* __restrict__ resid, const float* __restrict__ lns,
                           bf16_t* __restrict__ out, int n) {
  int i = blockIdx.x * blockDim.x + threadIdx.x;
  if (i >= n) return;
  int row = i >> 9;                       // / Dc
  out[i] = (bf16_t)(resid[i] / lns[row]);
}

__global__ void k_prep_up(const float* __restrict__ pf, const float* __restrict__ lns,
                          bf16_t* __restrict__ out, int n) {
  int i = blockIdx.x * blockDim.x + threadIdx.x;
  if (i >= n) return;
  int row = i / Fc;
  out[i] = (bf16_t)(pf[i] / lns[row]);
}

__global__ void k_convert(const float* __restrict__ in, bf16_t* __restrict__ out, int n) {
  int i = blockIdx.x * blockDim.x + threadIdx.x;
  if (i >= n) return;
  out[i] = (bf16_t)in[i];
}

// batched transpose+convert: out[b][c][r] = in[b][r][c]
__global__ void k_transpose_cb(const float* __restrict__ in, bf16_t* __restrict__ out,
                               int R, int C, int n) {
  int i = blockIdx.x * blockDim.x + threadIdx.x;
  if (i >= n) return;
  int rc = R * C;
  int b = i / rc;
  int rem = i - b * rc;
  int r = rem / C;
  int c = rem - r * C;
  out[(size_t)b * rc + (size_t)c * R + r] = (bf16_t)in[i];
}

// wvec[o] = sum_h sum_i (ub1[i]+ub2[i]) * W_OV[h,i,o]
__global__ void k_wvec(const float* __restrict__ W_OV, const float* __restrict__ ub1,
                       const float* __restrict__ ub2, float* __restrict__ wvec) {
  int o = blockIdx.x * blockDim.x + threadIdx.x;
  if (o >= Dc) return;
  float s = 0.f;
  for (int h = 0; h < Hc; ++h) {
    const float* W = W_OV + (size_t)h * Dc * Dc;
    for (int i = 0; i < Dc; ++i) s += (ub1[i] + ub2[i]) * W[(size_t)i * Dc + o];
  }
  wvec[o] = s;
}

// biasvec[f] = enc_w[f,:]·b_O + enc_b[f] - enc_w[f,:]·b_dec ; cvec[f] = enc_w[f,:]·wvec
__global__ void k_biascvec(const float* __restrict__ enc_w, const float* __restrict__ enc_b,
                           const float* __restrict__ b_O, const float* __restrict__ b_dec,
                           const float* __restrict__ wvec, float* __restrict__ biasvec,
                           float* __restrict__ cvec) {
  int f = blockIdx.x * blockDim.x + threadIdx.x;
  if (f >= Fc) return;
  const float* row = enc_w + (size_t)f * Dc;
  float s1 = 0.f, s2 = 0.f, s3 = 0.f;
  for (int i = 0; i < Dc; ++i) {
    float e = row[i];
    s1 += e * b_O[i];
    s2 += e * b_dec[i];
    s3 += e * wvec[i];
  }
  biasvec[f] = s1 + enc_b[f] - s2;
  cvec[f] = s3;
}

__global__ void k_init_approx(const float* __restrict__ lns, const float* __restrict__ biasvec,
                              const float* __restrict__ cvec, float* __restrict__ approx, int n) {
  int i = blockIdx.x * blockDim.x + threadIdx.x;
  if (i >= n) return;
  int row = i / Fc;
  int f = i - row * Fc;
  approx[i] = biasvec[f] + cvec[f] / lns[row];
}

// ======================= WMMA NT GEMM (async-LDS double buffered) =======================
// C[M,N] = A[M,K](row-major, lda) * B[N,K](row-major, ldb)^T
// MODE 0: store f32 C row-major (ldc)
// MODE 1: accumulate f32 C row-major (ldc)
// MODE 2: store bf16 C^T: out[n*ldc + m]   (ldc = leading dim of transposed array = M)
// MODE 3: (optional mask[m*ldc+n] multiply) store bf16 C row-major (ldc)
// Requires M%128==0, N%128==0, K%32==0 (and K>=64 for the pipeline; all call sites satisfy).

template <int MODE>
__global__ void __launch_bounds__(256)
gemm_nt(const bf16_t* __restrict__ A, int lda,
        const bf16_t* __restrict__ B, int ldb,
        void* __restrict__ Cv, int ldc,
        const float* __restrict__ mask,
        int M, int N, int K) {
  __shared__ alignas(16) bf16_t sA[2][128 * 40];   // 32 data + 8 pad bf16 per row
  __shared__ alignas(16) bf16_t sB[2][128 * 40];

  const int t = threadIdx.x;
  const int bm = blockIdx.y * 128;
  const int bn = blockIdx.x * 128;
  const int w = t >> 5;
  const int lane = t & 31;
  const int wm = (w >> 2) * 64;     // 2 m-strips of 64
  const int wn = (w & 3) * 32;      // 4 n-strips of 32
  const int m16 = lane & 15;
  const int hi = lane >> 4;

  v8f acc[4][2];
  for (int a = 0; a < 4; ++a)
    for (int b2 = 0; b2 < 2; ++b2)
      for (int r = 0; r < 8; ++r) acc[a][b2][r] = 0.f;

  // Staging assignment: 128 rows x 64B per tile = 512 16B chunks; 2 per thread.
  const int id0 = t * 2, id1 = id0 + 1;
  const int r0 = id0 >> 2, c0 = (id0 & 3) * 8;      // row, element offset (bf16)
  const int r1 = id1 >> 2, c1 = (id1 & 3) * 8;
  const bf16_t* gA0 = A + (size_t)(bm + r0) * lda + c0;
  const bf16_t* gA1 = A + (size_t)(bm + r1) * lda + c1;
  const bf16_t* gB0 = B + (size_t)(bn + r0) * ldb + c0;
  const bf16_t* gB1 = B + (size_t)(bn + r1) * ldb + c1;
  unsigned lA0[2], lA1[2], lB0[2], lB1[2];
#pragma unroll
  for (int bi = 0; bi < 2; ++bi) {
    lA0[bi] = lds_addr32(&sA[bi][r0 * 40 + c0]);
    lA1[bi] = lds_addr32(&sA[bi][r1 * 40 + c1]);
    lB0[bi] = lds_addr32(&sB[bi][r0 * 40 + c0]);
    lB1[bi] = lds_addr32(&sB[bi][r1 * 40 + c1]);
  }

  // 4 async global->LDS b128 per wave per tile-pair (tracked by ASYNCcnt)
  auto issue = [&](int k0, int bi) {
    asm volatile("global_load_async_to_lds_b128 %0, %1, off"
                 :: "v"(lA0[bi]), "v"((unsigned long long)(gA0 + k0)) : "memory");
    asm volatile("global_load_async_to_lds_b128 %0, %1, off"
                 :: "v"(lA1[bi]), "v"((unsigned long long)(gA1 + k0)) : "memory");
    asm volatile("global_load_async_to_lds_b128 %0, %1, off"
                 :: "v"(lB0[bi]), "v"((unsigned long long)(gB0 + k0)) : "memory");
    asm volatile("global_load_async_to_lds_b128 %0, %1, off"
                 :: "v"(lB1[bi]), "v"((unsigned long long)(gB1 + k0)) : "memory");
  };

  issue(0, 0);                       // prologue: fill buffer 0
  int bufi = 0;
  for (int k0 = 0; k0 < K; k0 += 32) {
    const bool more = (k0 + 32 < K);
    if (more) {
      issue(k0 + 32, bufi ^ 1);      // prefetch next tile into other buffer
      asm volatile("s_wait_asynccnt 4" ::: "memory");   // current tile's 4 done
    } else {
      asm volatile("s_wait_asynccnt 0" ::: "memory");
    }
    __syncthreads();                 // all waves' loads for bufi are in LDS

    const bf16_t* pAb = &sA[bufi][0];
    const bf16_t* pBb = &sB[bufi][0];

    // B fragments (32x16, K = hi*16 + 2v + j): one contiguous 32B read per lane
    FragBF bfrag[2];
#pragma unroll
    for (int nt = 0; nt < 2; ++nt) {
      const unsigned int* p =
          reinterpret_cast<const unsigned int*>(pBb + (wn + nt * 16 + m16) * 40);
      bfrag[nt].q[0] = *reinterpret_cast<const u32x4*>(p + hi * 8);
      bfrag[nt].q[1] = *reinterpret_cast<const u32x4*>(p + hi * 8 + 4);
    }
    // A fragments (16x32): u32 idx {0..7}+8*hi: two 16B reads per lane
#pragma unroll
    for (int mt = 0; mt < 4; ++mt) {
      FragBF af;
      const unsigned int* p =
          reinterpret_cast<const unsigned int*>(pAb + (wm + mt * 16 + m16) * 40);
      af.q[0] = *reinterpret_cast<const u32x4*>(p + hi * 8);
      af.q[1] = *reinterpret_cast<const u32x4*>(p + hi * 8 + 4);
#pragma unroll
      for (int nt = 0; nt < 2; ++nt) {
        acc[mt][nt] = __builtin_amdgcn_wmma_f32_16x16x32_bf16(
            false, af.v, false, bfrag[nt].v, (short)0, acc[mt][nt], false, false);
      }
    }
    __syncthreads();                 // compute on bufi done before it is refilled
    bufi ^= 1;
  }

  // epilogue: lane holds C rows r + 8*hi, col = m16, per 16x16 tile
#pragma unroll
  for (int mt = 0; mt < 4; ++mt)
#pragma unroll
    for (int nt = 0; nt < 2; ++nt)
#pragma unroll
      for (int r = 0; r < 8; ++r) {
        int row = bm + wm + mt * 16 + r + hi * 8;
        int col = bn + wn + nt * 16 + m16;
        float v = acc[mt][nt][r];
        if constexpr (MODE == 0) {
          ((float*)Cv)[(size_t)row * ldc + col] = v;
        } else if constexpr (MODE == 1) {
          float* C = (float*)Cv;
          size_t o = (size_t)row * ldc + col;
          C[o] = C[o] + v;
        } else if constexpr (MODE == 2) {
          ((bf16_t*)Cv)[(size_t)col * ldc + row] = (bf16_t)v;
        } else {
          if (mask) v *= mask[(size_t)row * ldc + col];
          ((bf16_t*)Cv)[(size_t)row * ldc + col] = (bf16_t)v;
        }
      }
}

// ======================= top-k + scatter + sparse decode =======================

__global__ void k_topk(const float* __restrict__ approx, const float* __restrict__ dec_w,
                       const float* __restrict__ b_dec, float* __restrict__ fb_out,
                       float* __restrict__ recon_out) {
  __shared__ float work[Fc];
  __shared__ float rval[256];
  __shared__ int ridx[256];
  __shared__ float selv[TOPK];
  __shared__ int seli[TOPK];

  const int row = blockIdx.x;
  const int t = threadIdx.x;
  const float* ap = approx + (size_t)row * Fc;
  for (int f = t; f < Fc; f += 256) work[f] = ap[f];
  __syncthreads();

  const float NEG = -3.402823466e38f;
  for (int it = 0; it < TOPK; ++it) {
    float bv = NEG;
    int bi = 0;
    for (int f = t; f < Fc; f += 256) {
      float v = work[f];
      if (v > bv) { bv = v; bi = f; }
    }
    rval[t] = bv; ridx[t] = bi;
    __syncthreads();
    for (int s = 128; s > 0; s >>= 1) {
      if (t < s) {
        if (rval[t + s] > rval[t] ||
            (rval[t + s] == rval[t] && ridx[t + s] < ridx[t])) {
          rval[t] = rval[t + s];
          ridx[t] = ridx[t + s];
        }
      }
      __syncthreads();
    }
    if (t == 0) {
      selv[it] = rval[0];
      seli[it] = ridx[0];
      work[ridx[0]] = NEG;
    }
    __syncthreads();
  }

  float* fb = fb_out + (size_t)row * Fc;
  for (int f = t; f < Fc; f += 256) fb[f] = 0.f;
  __syncthreads();
  if (t < TOPK) fb[seli[t]] = fmaxf(selv[t], 0.f);

  for (int d = t; d < Dc; d += 256) {
    float s = b_dec[d];
    const float* dw = dec_w + (size_t)d * Fc;
    for (int j = 0; j < TOPK; ++j) s += fmaxf(selv[j], 0.f) * dw[seli[j]];
    recon_out[(size_t)row * Dc + d] = s;
  }
}

// ======================= host orchestration =======================

extern "C" void kernel_launch(void* const* d_in, const int* in_sizes, int n_in,
                              void* d_out, int out_size, void* d_ws, size_t ws_size,
                              hipStream_t stream) {
  (void)in_sizes; (void)n_in; (void)out_size; (void)ws_size;
  const float* resid = (const float*)d_in[0];
  const float* lns   = (const float*)d_in[1];
  const float* probs = (const float*)d_in[2];
  const float* W_OV  = (const float*)d_in[3];
  const float* b_O   = (const float*)d_in[4];
  const float* enc_w = (const float*)d_in[5];
  const float* enc_b = (const float*)d_in[6];
  const float* b_dec = (const float*)d_in[7];
  const float* dec_w = (const float*)d_in[8];
  const float* up_dec[2]  = {(const float*)d_in[9],  (const float*)d_in[13]};
  const float* up_bdec[2] = {(const float*)d_in[10], (const float*)d_in[14]};
  const float* pf[2]      = {(const float*)d_in[11], (const float*)d_in[15]};
  const float* maskp[2]   = {(const float*)d_in[12], (const float*)d_in[16]};

  char* ws = (char*)d_ws;
  size_t off = 0;
  auto take = [&](size_t bytes) -> char* {
    char* p = ws + off;
    off = (off + bytes + 255) & ~(size_t)255;
    return p;
  };
  bf16_t* act_bf     = (bf16_t*)take((size_t)BS * Dc * 2);            // [2048,512]
  bf16_t* probs_bf   = (bf16_t*)take((size_t)Bc * Hc * Sc * Sc * 2);  // [b,h,S,S]
  bf16_t* encw_bf    = (bf16_t*)take((size_t)Fc * Dc * 2);            // [F,D]
  bf16_t* wovt_bf    = (bf16_t*)take((size_t)Hc * Dc * Dc * 2);       // [h][o][i]
  bf16_t* ov_bf      = (bf16_t*)take((size_t)BS * Dc * 2);            // [2048,512]
  bf16_t* preT_bf    = (bf16_t*)take((size_t)Fc * BS * 2);            // [F,2048]
  float*  approx     = (float*)take((size_t)BS * Fc * 4);             // [b,s,F]
  bf16_t* updecT_bf  = (bf16_t*)take((size_t)Fc * Dc * 2);            // [F,D]
  bf16_t* uppost_bf  = (bf16_t*)take((size_t)BS * Fc * 2);            // [b,s,F]
  bf16_t* tmpT_bf    = (bf16_t*)take((size_t)Fc * Dc * 2);            // [u,o]
  bf16_t* vw_bf      = (bf16_t*)take((size_t)Fc * Fc * 2);            // [d,u]
  bf16_t* postovT_bf = (bf16_t*)take((size_t)Fc * Sc * 2);            // [d,k]
  float*  wvec       = (float*)take((size_t)Dc * 4);
  float*  biasvec    = (float*)take((size_t)Fc * 4);
  float*  cvec       = (float*)take((size_t)Fc * 4);

  const int th = 256;
  k_prep_act<<<(BS * Dc) / th, th, 0, stream>>>(resid, lns, act_bf, BS * Dc);
  k_convert<<<(Bc * Hc * Sc * Sc) / th, th, 0, stream>>>(probs, probs_bf, Bc * Hc * Sc * Sc);
  k_convert<<<(Fc * Dc) / th, th, 0, stream>>>(enc_w, encw_bf, Fc * Dc);
  k_transpose_cb<<<(Hc * Dc * Dc) / th, th, 0, stream>>>(W_OV, wovt_bf, Dc, Dc, Hc * Dc * Dc);
  k_wvec<<<(Dc + th - 1) / th, th, 0, stream>>>(W_OV, up_bdec[0], up_bdec[1], wvec);
  k_biascvec<<<Fc / th, th, 0, stream>>>(enc_w, enc_b, b_O, b_dec, wvec, biasvec, cvec);
  k_init_approx<<<(BS * Fc) / th, th, 0, stream>>>(lns, biasvec, cvec, approx, BS * Fc);

  // ---- initial contributions: approx += probs · (act·W_OV·enc_w^T), summed over h ----
  for (int h = 0; h < Hc; ++h) {
    // ov_h[2048,512] = act[2048,512] · W_OVt[h][512,512]^T
    gemm_nt<3><<<dim3(Dc / 128, BS / 128), 256, 0, stream>>>(
        act_bf, Dc, wovt_bf + (size_t)h * Dc * Dc, Dc, ov_bf, Dc, nullptr, BS, Dc, Dc);
    // preT[F,2048] = (ov_h[2048,512] · enc_w[F,512]^T)^T
    gemm_nt<2><<<dim3(Fc / 128, BS / 128), 256, 0, stream>>>(
        ov_bf, Dc, encw_bf, Dc, preT_bf, BS, nullptr, BS, Fc, Dc);
    for (int b = 0; b < Bc; ++b) {
      // approx[b] += probs[b,h][S,S] · preT[:, b*S : b*S+S]^T
      gemm_nt<1><<<dim3(Fc / 128, Sc / 128), 256, 0, stream>>>(
          probs_bf + ((size_t)(b * Hc + h)) * Sc * Sc, Sc,
          preT_bf + (size_t)b * Sc, BS,
          approx + (size_t)b * Sc * Fc, Fc, nullptr, Sc, Fc, Sc);
    }
  }

  // ---- upstream pruned contributions ----
  for (int u = 0; u < 2; ++u) {
    k_transpose_cb<<<(Dc * Fc) / th, th, 0, stream>>>(up_dec[u], updecT_bf, Dc, Fc, Dc * Fc);
    k_prep_up<<<(BS * Fc) / th, th, 0, stream>>>(pf[u], lns, uppost_bf, BS * Fc);
    for (int h = 0; h < Hc; ++h) {
      // tmpT[u_feat, o] = up_decT[F,512] · W_OVt[h][512,512]^T
      gemm_nt<3><<<dim3(Dc / 128, Fc / 128), 256, 0, stream>>>(
          updecT_bf, Dc, wovt_bf + (size_t)h * Dc * Dc, Dc, tmpT_bf, Dc, nullptr,
          Fc, Dc, Dc);
      // vw[d,u] = (enc_w[F,512] · tmpT[F,512]^T) * mask[d,u]
      gemm_nt<3><<<dim3(Fc / 128, Fc / 128), 256, 0, stream>>>(
          encw_bf, Dc, tmpT_bf, Dc, vw_bf, Fc, maskp[u], Fc, Fc, Dc);
      for (int b = 0; b < Bc; ++b) {
        // postovT[d,k] = (up_post[b][S,F] · vw[F,F]^T)^T
        gemm_nt<2><<<dim3(Fc / 128, Sc / 128), 256, 0, stream>>>(
            uppost_bf + (size_t)b * Sc * Fc, Fc, vw_bf, Fc, postovT_bf, Sc, nullptr,
            Sc, Fc, Fc);
        // approx[b] += probs[b,h][S,S] · postovT[F,S]^T
        gemm_nt<1><<<dim3(Fc / 128, Sc / 128), 256, 0, stream>>>(
            probs_bf + ((size_t)(b * Hc + h)) * Sc * Sc, Sc, postovT_bf, Sc,
            approx + (size_t)b * Sc * Fc, Fc, nullptr, Sc, Fc, Sc);
      }
    }
  }

  // ---- topk + relu + scatter + sparse decode ----
  float* fb_out = (float*)d_out;
  float* recon_out = fb_out + (size_t)BS * Fc;
  k_topk<<<BS, 256, 0, stream>>>(approx, dec_w, b_dec, fb_out, recon_out);
}
